// AutoEncoderDynamicTopK_59416577573585
// MI455X (gfx1250) — matrix operand order repaired and checked by reading, not compiled, not run on hardware
//
#include <hip/hip_runtime.h>

// ---------------------------------------------------------------------------
// AutoEncoder with dynamic per-row top-k, MI455X (gfx1250, wave32, WMMA).
// Pipeline:
//  [0] split x-b_dec and W_enc into bf16 hi/lo planes (one pass, BW-trivial)
//  [1] encoder GEMM: 3-product bf16 split WMMA (~fp32 accuracy), double-
//      buffered LDS tiles filled via GLOBAL_LOAD_ASYNC_TO_LDS_B128 (ASYNCcnt)
//  [2] per-row radix-select top-k (stable ties) -> sparse lists
//  [3] transpose W_dec -> Wt[F,D] (reuses dead W_enc-split region)
//  [4] sparse decode: out = b_dec + sum_j val_j * Wt[idx_j,:]  (fp32 exact)
// ---------------------------------------------------------------------------

#define BB 4096
#define DD 2048
#define FF 16384
#define MAXK 320

typedef __bf16 bf16x16 __attribute__((ext_vector_type(16)));
typedef float  f32x8   __attribute__((ext_vector_type(8)));
typedef int    v4i     __attribute__((ext_vector_type(4)));

union FragU { unsigned int u[8]; bf16x16 v; };
union BfU { __bf16 b; unsigned short u; };

// ---- CDNA5 async global->LDS copy (ASYNCcnt) with portable fallback -------
#if defined(__has_builtin)
#if __has_builtin(__builtin_amdgcn_global_load_async_to_lds_b128)
#define HAVE_ASYNC_LDS 1
#endif
#if __has_builtin(__builtin_amdgcn_s_wait_asynccnt)
#define HAVE_WAIT_ASYNC 1
#endif
#endif

__device__ __forceinline__ void copy16_g2l(const void* g, void* l) {
#if defined(HAVE_ASYNC_LDS)
  typedef __attribute__((address_space(1))) v4i* gp_t;   // global v4i*
  typedef __attribute__((address_space(3))) v4i* lp_t;   // LDS v4i*
  __builtin_amdgcn_global_load_async_to_lds_b128((gp_t)g, (lp_t)l, 0, 0);
#else
  *(uint4*)l = *(const uint4*)g;
#endif
}
__device__ __forceinline__ void wait_async_copies() {
#if defined(HAVE_WAIT_ASYNC)
  __builtin_amdgcn_s_wait_asynccnt((short)0);
#elif defined(HAVE_ASYNC_LDS)
  asm volatile("s_wait_asynccnt 0x0" ::: "memory");
#endif
}

// ---------------------------------------------------------------------------
// Kernel 0: elementwise split to bf16 hi/lo planes.  in is [R, DD] row-major,
//           optional per-column subtrahend (b_dec) of length DD.
// ---------------------------------------------------------------------------
__global__ __launch_bounds__(256) void split_bf16_kernel(
    const float* __restrict__ in, const float* __restrict__ sub,
    unsigned short* __restrict__ hi, unsigned short* __restrict__ lo) {
  const size_t idx = ((size_t)blockIdx.x * 256 + threadIdx.x) * 4;
  const int col = (int)(idx & (DD - 1));
  float4 v = *(const float4*)(in + idx);
  if (sub) {
    const float4 s = *(const float4*)(sub + col);
    v.x -= s.x; v.y -= s.y; v.z -= s.z; v.w -= s.w;
  }
  BfU h0, h1, h2, h3, l0, l1, l2, l3;
  h0.b = (__bf16)v.x; l0.b = (__bf16)(v.x - (float)h0.b);
  h1.b = (__bf16)v.y; l1.b = (__bf16)(v.y - (float)h1.b);
  h2.b = (__bf16)v.z; l2.b = (__bf16)(v.z - (float)h2.b);
  h3.b = (__bf16)v.w; l3.b = (__bf16)(v.w - (float)h3.b);
  uint2 hp, lp;
  hp.x = (unsigned int)h0.u | ((unsigned int)h1.u << 16);
  hp.y = (unsigned int)h2.u | ((unsigned int)h3.u << 16);
  lp.x = (unsigned int)l0.u | ((unsigned int)l1.u << 16);
  lp.y = (unsigned int)l2.u | ((unsigned int)l3.u << 16);
  *(uint2*)(hi + idx) = hp;
  *(uint2*)(lo + idx) = lp;
}

// ---------------------------------------------------------------------------
// Kernel 1: acts = relu((x - b_dec) @ W_enc^T + b_enc)    [B,F] fp32
//   M=B, N=F, K=D.  WG tile 128x128, K-tile 32, double-buffered LDS.
//   8 waves as 2(M) x 4(N); wave tile 64x32 => 4x2 subtiles, 24 WMMA/K-step.
// ---------------------------------------------------------------------------
__global__ __launch_bounds__(256) void enc_gemm_kernel(
    const unsigned short* __restrict__ xh, const unsigned short* __restrict__ xl,
    const unsigned short* __restrict__ wh, const unsigned short* __restrict__ wl,
    const float* __restrict__ b_enc, float* __restrict__ acts) {
  // bf16 pairs packed as uint32: [buf][row][k-pair 0..15]
  __shared__ unsigned int sAh[2][128][16];
  __shared__ unsigned int sAl[2][128][16];
  __shared__ unsigned int sBh[2][128][16];
  __shared__ unsigned int sBl[2][128][16];

  const int tid = threadIdx.x;
  const int m0 = blockIdx.y * 128;
  const int n0 = blockIdx.x * 128;
  const int wid = tid >> 5;
  const int lane = tid & 31;
  const int wm = wid >> 2;        // 0..1  (64-row slice)
  const int wn = wid & 3;         // 0..3  (32-col slice)
  const int h = lane >> 4;
  const int l15 = lane & 15;

  f32x8 acc[4][2];
#pragma unroll
  for (int i = 0; i < 4; ++i)
#pragma unroll
    for (int j = 0; j < 2; ++j)
#pragma unroll
      for (int r = 0; r < 8; ++r) acc[i][j][r] = 0.0f;

  const int KT = DD / 32;

  // tile copier: 512 16B-chunks per plane tile -> 2 chunks/thread/plane
  auto load_tiles = [&](int buf, int kb) {
#pragma unroll
    for (int i = 0; i < 2; ++i) {
      const int lin = tid + i * 256;   // 0..511
      const int row = lin >> 2;        // 0..127
      const int c4 = lin & 3;          // 16B chunk within 32-k row (8 bf16)
      const size_t ga = (size_t)(m0 + row) * DD + kb + c4 * 8;
      const size_t gb = (size_t)(n0 + row) * DD + kb + c4 * 8;
      copy16_g2l(xh + ga, &sAh[buf][row][c4 * 4]);
      copy16_g2l(xl + ga, &sAl[buf][row][c4 * 4]);
      copy16_g2l(wh + gb, &sBh[buf][row][c4 * 4]);
      copy16_g2l(wl + gb, &sBl[buf][row][c4 * 4]);
    }
  };

  load_tiles(0, 0);
  wait_async_copies();
  __syncthreads();

  for (int kt = 0; kt < KT; ++kt) {
    const int cur = kt & 1;
    if (kt + 1 < KT) load_tiles(cur ^ 1, (kt + 1) * 32);

    // ---- fragments per ISA VGPR layouts ----
    FragU ah[4], al[4], bh[2], bl[2];
#pragma unroll
    for (int i = 0; i < 4; ++i) {
      const int ml = wm * 64 + i * 16 + l15;
#pragma unroll
      for (int v = 0; v < 8; ++v) {
        // A 16x32 bf16: lanes0-15 K in {0..7,16..23}; lanes16-31 {8..15,24..31}
        const int t = v + (v & 4) + (h << 2);
        ah[i].u[v] = sAh[cur][ml][t];
        al[i].u[v] = sAl[cur][ml][t];
      }
    }
#pragma unroll
    for (int j = 0; j < 2; ++j) {
      const int nl = wn * 32 + j * 16 + l15;
#pragma unroll
      for (int v = 0; v < 8; ++v) {
        // B 32x16 bf16: lane = N; lanes0-15 K in {0..7,16..23}; lanes16-31 {8..15,24..31}
        const int t = v + (v & 4) + (h << 2);
        bh[j].u[v] = sBh[cur][nl][t];
        bl[j].u[v] = sBl[cur][nl][t];
      }
    }

    // ---- acc += Alo*Bhi + Ahi*Blo + Ahi*Bhi  (bf16 split, ~fp32) ----
#pragma unroll
    for (int i = 0; i < 4; ++i)
#pragma unroll
      for (int j = 0; j < 2; ++j) {
        acc[i][j] = __builtin_amdgcn_wmma_f32_16x16x32_bf16(
            false, al[i].v, false, bh[j].v, (short)0, acc[i][j], false, false);
        acc[i][j] = __builtin_amdgcn_wmma_f32_16x16x32_bf16(
            false, ah[i].v, false, bl[j].v, (short)0, acc[i][j], false, false);
        acc[i][j] = __builtin_amdgcn_wmma_f32_16x16x32_bf16(
            false, ah[i].v, false, bh[j].v, (short)0, acc[i][j], false, false);
      }

    wait_async_copies();   // next buffer fully in LDS
    __syncthreads();
  }

  // ---- epilogue: +b_enc, relu, store fp32 acts ----
#pragma unroll
  for (int i = 0; i < 4; ++i)
#pragma unroll
    for (int j = 0; j < 2; ++j) {
      const int gn = n0 + wn * 32 + j * 16 + l15;
      const float bias = b_enc[gn];
      const int gmb = m0 + wm * 64 + i * 16 + h * 8;  // C/D: vgpr r -> M=r+8h
#pragma unroll
      for (int r = 0; r < 8; ++r) {
        float val = acc[i][j][r] + bias;
        val = val > 0.0f ? val : 0.0f;
        acts[(size_t)(gmb + r) * FF + gn] = val;
      }
    }
}

// ---------------------------------------------------------------------------
// Kernel 2: per-row top-k via 4-pass radix select on float bits (acts >= 0),
//           stable compaction (ties kept by ascending index, like argsort).
// ---------------------------------------------------------------------------
__global__ __launch_bounds__(256) void topk_kernel(
    const float* __restrict__ acts, const int* __restrict__ kvals,
    int* __restrict__ sp_idx, float* __restrict__ sp_val, int* __restrict__ sp_cnt) {
  const int b = blockIdx.x;
  const int tid = threadIdx.x;
  const unsigned int* row = (const unsigned int*)(acts + (size_t)b * FF);

  __shared__ unsigned int hist[256];
  __shared__ unsigned int eqoff[257];
  __shared__ unsigned int sh_byte, sh_rem, gt_ctr;

  int k = kvals[b];
  if (k <= 0) { if (tid == 0) sp_cnt[b] = 0; return; }
  if (k > MAXK) k = MAXK;

  unsigned int prefix = 0;
  unsigned int remaining = (unsigned int)k;
#pragma unroll
  for (int pass = 0; pass < 4; ++pass) {
    const int shift = 24 - 8 * pass;
    const unsigned int himask = (pass == 0) ? 0u : ~((1u << (shift + 8)) - 1u);
    hist[tid] = 0u;
    __syncthreads();
    for (int idx = tid; idx < FF; idx += 256) {
      unsigned int v = row[idx];
      if ((v & himask) == prefix) atomicAdd(&hist[(v >> shift) & 255u], 1u);
    }
    __syncthreads();
    if (tid == 0) {
      unsigned int r = remaining, byte = 0;
      for (int bin = 255; bin >= 0; --bin) {
        unsigned int c = hist[bin];
        if (c >= r) { byte = (unsigned int)bin; break; }
        r -= c;
      }
      sh_byte = byte; sh_rem = r;
    }
    __syncthreads();
    prefix |= (sh_byte << shift);
    remaining = sh_rem;
    __syncthreads();
  }

  const unsigned int T = prefix;                 // bit pattern of k-th largest
  const unsigned int count_gt = (unsigned int)k - remaining;

  const int base = tid * (FF / 256);
  unsigned int eqc = 0;
  for (int q = 0; q < FF / 256; ++q) eqc += (row[base + q] == T) ? 1u : 0u;
  eqoff[tid] = eqc;
  if (tid == 0) gt_ctr = 0u;
  __syncthreads();
  if (tid == 0) {
    unsigned int s = 0;
    for (int t2 = 0; t2 < 256; ++t2) { unsigned int c = eqoff[t2]; eqoff[t2] = s; s += c; }
    eqoff[256] = s;
  }
  __syncthreads();

  unsigned int myeq = eqoff[tid];
  for (int q = 0; q < FF / 256; ++q) {
    const int idx = base + q;
    const unsigned int v = row[idx];
    if (v > T) {
      unsigned int s = atomicAdd(&gt_ctr, 1u);
      if (s < (unsigned int)k) {
        sp_idx[(size_t)b * MAXK + s] = idx;
        sp_val[(size_t)b * MAXK + s] = __uint_as_float(v);
      }
    } else if (v == T) {
      if (myeq < remaining) {
        unsigned int s = count_gt + myeq;
        sp_idx[(size_t)b * MAXK + s] = idx;
        sp_val[(size_t)b * MAXK + s] = __uint_as_float(v);
      }
      ++myeq;
    }
  }
  if (tid == 0) {
    unsigned int tot_eq = eqoff[256];
    unsigned int kept_eq = remaining < tot_eq ? remaining : tot_eq;
    sp_cnt[b] = (int)(count_gt + kept_eq);
  }
}

// ---------------------------------------------------------------------------
// Kernel 3: Wt[F,D] = transpose(W_dec[D,F])
// ---------------------------------------------------------------------------
__global__ __launch_bounds__(256) void transpose_kernel(
    const float* __restrict__ in, float* __restrict__ out) {
  __shared__ float tile[32][33];
  const int f0 = blockIdx.x * 32;
  const int d0 = blockIdx.y * 32;
  const int tx = threadIdx.x & 31;
  const int ty = threadIdx.x >> 5;  // 0..7
#pragma unroll
  for (int i = 0; i < 4; ++i)
    tile[ty + i * 8][tx] = in[(size_t)(d0 + ty + i * 8) * FF + f0 + tx];
  __syncthreads();
#pragma unroll
  for (int i = 0; i < 4; ++i)
    out[(size_t)(f0 + ty + i * 8) * DD + d0 + tx] = tile[tx][ty + i * 8];
}

// ---------------------------------------------------------------------------
// Kernel 4: out[b,:] = b_dec + sum_j val_j * Wt[idx_j, :]   (k-sparse decode)
// ---------------------------------------------------------------------------
__global__ __launch_bounds__(256) void decode_kernel(
    const float* __restrict__ Wt, const int* __restrict__ sp_idx,
    const float* __restrict__ sp_val, const int* __restrict__ sp_cnt,
    const float* __restrict__ b_dec, float* __restrict__ out) {
  const int b = blockIdx.x;
  const int tid = threadIdx.x;
  __shared__ int s_idx[MAXK];
  __shared__ float s_val[MAXK];
  const int cnt = sp_cnt[b];
  for (int j = tid; j < cnt; j += 256) {
    s_idx[j] = sp_idx[(size_t)b * MAXK + j];
    s_val[j] = sp_val[(size_t)b * MAXK + j];
  }
  __syncthreads();

  float4 a0 = ((const float4*)b_dec)[tid * 2];
  float4 a1 = ((const float4*)b_dec)[tid * 2 + 1];
  for (int j = 0; j < cnt; ++j) {
    const float v = s_val[j];
    const float4* wr = (const float4*)(Wt + (size_t)s_idx[j] * DD);
    const float4 w0 = wr[tid * 2];
    const float4 w1 = wr[tid * 2 + 1];
    a0.x = fmaf(v, w0.x, a0.x); a0.y = fmaf(v, w0.y, a0.y);
    a0.z = fmaf(v, w0.z, a0.z); a0.w = fmaf(v, w0.w, a0.w);
    a1.x = fmaf(v, w1.x, a1.x); a1.y = fmaf(v, w1.y, a1.y);
    a1.z = fmaf(v, w1.z, a1.z); a1.w = fmaf(v, w1.w, a1.w);
  }
  float4* o = (float4*)(out + (size_t)b * DD);
  o[tid * 2] = a0;
  o[tid * 2 + 1] = a1;
}

// ---------------------------------------------------------------------------
extern "C" void kernel_launch(void* const* d_in, const int* in_sizes, int n_in,
                              void* d_out, int out_size, void* d_ws, size_t ws_size,
                              hipStream_t stream) {
  (void)in_sizes; (void)n_in; (void)out_size; (void)ws_size;
  const float* x     = (const float*)d_in[0];   // [B,D]
  const int*   kvals = (const int*)d_in[1];     // [B]
  const float* W_enc = (const float*)d_in[2];   // [F,D]
  const float* b_enc = (const float*)d_in[3];   // [F]
  const float* W_dec = (const float*)d_in[4];   // [D,F]
  const float* b_dec = (const float*)d_in[5];   // [D]
  float* out = (float*)d_out;                   // [B,D]

  char* ws = (char*)d_ws;
  size_t off = 0;
  float* acts = (float*)(ws + off); off += (size_t)BB * FF * 4;      // 256MB
  const size_t we_off = off;                                         // Wt aliases here
  unsigned short* We_h = (unsigned short*)(ws + off); off += (size_t)FF * DD * 2;  // 64MB
  unsigned short* We_l = (unsigned short*)(ws + off); off += (size_t)FF * DD * 2;  // 64MB
  unsigned short* xm_h = (unsigned short*)(ws + off); off += (size_t)BB * DD * 2;  // 16MB
  unsigned short* xm_l = (unsigned short*)(ws + off); off += (size_t)BB * DD * 2;  // 16MB
  int*   sp_idx = (int*)(ws + off);   off += (size_t)BB * MAXK * 4;
  float* sp_val = (float*)(ws + off); off += (size_t)BB * MAXK * 4;
  int*   sp_cnt = (int*)(ws + off);                                  // ~426MB total
  float* Wt = (float*)(ws + we_off);   // [F,D] fp32, reuses dead We_h/We_l

  // [0] precompute bf16 hi/lo planes (kills per-tile conversion: x was split
  //     256x and W_enc 32x inside the old GEMM loop)
  split_bf16_kernel<<<dim3((unsigned)((size_t)BB * DD / 1024)), 256, 0, stream>>>(
      x, b_dec, xm_h, xm_l);
  split_bf16_kernel<<<dim3((unsigned)((size_t)FF * DD / 1024)), 256, 0, stream>>>(
      W_enc, nullptr, We_h, We_l);
  // [1] WMMA encoder GEMM
  enc_gemm_kernel<<<dim3(FF / 128, BB / 128), 256, 0, stream>>>(
      xm_h, xm_l, We_h, We_l, b_enc, acts);
  // [2] per-row dynamic top-k
  topk_kernel<<<dim3(BB), 256, 0, stream>>>(acts, kvals, sp_idx, sp_val, sp_cnt);
  // [3] W_dec transpose into region freed by the W_enc splits
  transpose_kernel<<<dim3(FF / 32, DD / 32), 256, 0, stream>>>(W_dec, Wt);
  // [4] k-sparse decode
  decode_kernel<<<dim3(BB), 256, 0, stream>>>(Wt, sp_idx, sp_val, sp_cnt, b_dec, out);
}
